// TriangleMultiplicationOutgoing_24438363914905
// MI455X (gfx1250) — matrix operand order repaired and checked by reading, not compiled, not run on hardware
//
#include <hip/hip_runtime.h>
#include <hip/hip_bf16.h>

// ---------------------------------------------------------------------------
// Triangle Multiplication (Outgoing) for MI455X / gfx1250, wave32 + WMMA.
// GEMMs use v_wmma_f32_16x16x32_f16 (f16 operands, f32 accumulate).
// Triangle einsum stages tiles in LDS with GLOBAL_LOAD_ASYNC_TO_LDS_B128
// (ASYNCcnt) and double-buffers the k-loop.
// ---------------------------------------------------------------------------

typedef __attribute__((ext_vector_type(16))) _Float16 v16h;
typedef __attribute__((ext_vector_type(8)))  _Float16 v8h;
typedef __attribute__((ext_vector_type(4)))  _Float16 v4h;
typedef __attribute__((ext_vector_type(8)))  float    v8f;
typedef int aint4 __attribute__((vector_size(16)));   // b128 payload type

#define NDIM 512
#define EDIM 128
#define MROWS (NDIM * NDIM)          // 262144
#define NN    (NDIM * NDIM)          // per-channel matrix size

#if defined(__has_builtin)
#if __has_builtin(__builtin_amdgcn_global_load_async_to_lds_b128)
#define HAS_ASYNC_LDS 1
#endif
#if __has_builtin(__builtin_amdgcn_s_wait_asynccnt)
#define HAS_WAIT_ASYNC 1
#endif
#endif

__device__ __forceinline__ v8f zero8f() {
    v8f z;
#pragma unroll
    for (int i = 0; i < 8; ++i) z[i] = 0.0f;
    return z;
}

__device__ __forceinline__ float sigmoidf_(float x) {
    return 1.0f / (1.0f + __expf(-x));
}

// A fragment: 16(M) x 32(K) f16, source row-major [row][k], leading dim ld.
// ISA layout: lanes 0-15 row=lane, K 0..7 / 16..23; lanes 16-31 row=lane-16,
// K 8..15 / 24..31.
__device__ __forceinline__ v16h load_a_frag(const _Float16* __restrict__ base,
                                            int row0, int k0, int ld, int lane) {
    int r  = lane & 15;
    int hi = lane >> 4;
    const _Float16* p = base + (size_t)(row0 + r) * ld + k0 + hi * 8;
    v8h lo  = *(const v8h*)p;
    v8h hi8 = *(const v8h*)(p + 16);
    return __builtin_shufflevector(lo, hi8, 0, 1, 2, 3, 4, 5, 6, 7,
                                            8, 9, 10, 11, 12, 13, 14, 15);
}

// B fragment: 32(K) x 16(N) f16, sourced from B^T stored row-major [n][k].
// ISA layout: lanes 0-15 col=lane K 0..7 / 16..23; lanes 16-31 col=lane-16,
// K 8..15 / 24..31.
__device__ __forceinline__ v16h load_b_frag(const _Float16* __restrict__ baseT,
                                            int col0, int k0, int ld, int lane) {
    int n  = lane & 15;
    int kh = lane >> 4;
    const _Float16* p = baseT + (size_t)(col0 + n) * ld + k0 + kh * 8;
    v8h lo  = *(const v8h*)p;
    v8h hi8 = *(const v8h*)(p + 16);
    return __builtin_shufflevector(lo, hi8, 0, 1, 2, 3, 4, 5, 6, 7,
                                            8, 9, 10, 11, 12, 13, 14, 15);
}

__device__ __forceinline__ v8f wmma16(v16h a, v16h b, v8f c) {
    return __builtin_amdgcn_wmma_f32_16x16x32_f16(false, a, false, b,
                                                  (short)0, c, false, false);
}

// 16-byte global -> LDS copy (async DMA when available).
__device__ __forceinline__ void async_copy16(const _Float16* __restrict__ g,
                                             _Float16* l) {
#if defined(HAS_ASYNC_LDS)
    __builtin_amdgcn_global_load_async_to_lds_b128(
        (__attribute__((address_space(1))) aint4*)g,
        (__attribute__((address_space(3))) aint4*)l, 0, 0);
#else
    *(v8h*)l = *(const v8h*)g;
#endif
}

__device__ __forceinline__ void wait_async_zero() {
#if defined(HAS_WAIT_ASYNC)
    __builtin_amdgcn_s_wait_asynccnt(0);
#elif defined(HAS_ASYNC_LDS)
    asm volatile("s_wait_asynccnt 0x0" ::: "memory");
#endif
}

// ---------------------------------------------------------------------------
// Weight prep: wt[c][e] = (f16) w[e][c]   (128x128 transpose + convert)
// ---------------------------------------------------------------------------
__global__ void wprep_kernel(const float* __restrict__ w, _Float16* __restrict__ wt) {
    int idx = blockIdx.x * blockDim.x + threadIdx.x;   // 0..16383
    int c = idx >> 7;
    int e = idx & 127;
    wt[idx] = (_Float16)w[(size_t)e * EDIM + c];
}

// ---------------------------------------------------------------------------
// Row LayerNorm over 128 elements, f32 in -> f16 out. One wave per row.
// ---------------------------------------------------------------------------
__global__ void ln_rows_kernel(const float* __restrict__ x,
                               const float* __restrict__ gamma,
                               const float* __restrict__ beta,
                               _Float16* __restrict__ out, int nrows) {
    int wid  = threadIdx.x >> 5;
    int lane = threadIdx.x & 31;
    int row  = blockIdx.x * (blockDim.x >> 5) + wid;
    if (row >= nrows) return;
    const float4 v = ((const float4*)(x + (size_t)row * EDIM))[lane];
    float s  = v.x + v.y + v.z + v.w;
    float s2 = v.x * v.x + v.y * v.y + v.z * v.z + v.w * v.w;
#pragma unroll
    for (int off = 16; off > 0; off >>= 1) {
        s  += __shfl_xor(s, off, 32);
        s2 += __shfl_xor(s2, off, 32);
    }
    float mean = s * (1.0f / 128.0f);
    float var  = s2 * (1.0f / 128.0f) - mean * mean;
    float rs   = rsqrtf(var + 1e-5f);
    const float4 gv = ((const float4*)gamma)[lane];
    const float4 bv = ((const float4*)beta)[lane];
    v4h o;
    o[0] = (_Float16)((v.x - mean) * rs * gv.x + bv.x);
    o[1] = (_Float16)((v.y - mean) * rs * gv.y + bv.y);
    o[2] = (_Float16)((v.z - mean) * rs * gv.z + bv.z);
    o[3] = (_Float16)((v.w - mean) * rs * gv.w + bv.w);
    *(v4h*)(out + (size_t)row * EDIM + lane * 4) = o;
}

// ---------------------------------------------------------------------------
// Fused projection pair:  out = sigmoid(z@gw + gb) * (z@lw + lb)
// Output channel-major: outT[c][i][k] f16, m = i*512 + k.
// Block 128 (4 waves); wave computes 16m x 32c. Grid (M/64, 128/32).
// ---------------------------------------------------------------------------
__global__ void proj_pair_kernel(const _Float16* __restrict__ z,
                                 const _Float16* __restrict__ gwt,
                                 const float*    __restrict__ gb,
                                 const _Float16* __restrict__ lwt,
                                 const float*    __restrict__ lb,
                                 _Float16* __restrict__ outT) {
    int lane = threadIdx.x & 31;
    int w    = threadIdx.x >> 5;
    int m0   = blockIdx.x * 64 + w * 16;
    int c0   = blockIdx.y * 32;

    v8f ag[2], al[2];
#pragma unroll
    for (int t = 0; t < 2; ++t) { ag[t] = zero8f(); al[t] = zero8f(); }

#pragma unroll
    for (int kk = 0; kk < EDIM; kk += 32) {
        v16h af = load_a_frag(z, m0, kk, EDIM, lane);
#pragma unroll
        for (int t = 0; t < 2; ++t) {
            v16h bg = load_b_frag(gwt, c0 + t * 16, kk, EDIM, lane);
            ag[t] = wmma16(af, bg, ag[t]);
            v16h bl = load_b_frag(lwt, c0 + t * 16, kk, EDIM, lane);
            al[t] = wmma16(af, bl, al[t]);
        }
    }

    int i    = m0 >> 9;
    int kpos = (m0 & (NDIM - 1)) + (lane >> 4) * 8;
#pragma unroll
    for (int t = 0; t < 2; ++t) {
        int   c  = c0 + t * 16 + (lane & 15);
        float gB = gb[c];
        float lB = lb[c];
        v8h pack;
#pragma unroll
        for (int r = 0; r < 8; ++r) {
            float gvv = ag[t][r] + gB;
            float lvv = al[t][r] + lB;
            pack[r] = (_Float16)(sigmoidf_(gvv) * lvv);
        }
        *(v8h*)(outT + ((size_t)c << 18) + ((size_t)i << 9) + kpos) = pack;
    }
}

// ---------------------------------------------------------------------------
// Gate projection: out[m][c] = sigmoid(z@w + b), f16 row-major output.
// ---------------------------------------------------------------------------
__global__ void proj_g_kernel(const _Float16* __restrict__ z,
                              const _Float16* __restrict__ wt,
                              const float*    __restrict__ bias,
                              _Float16* __restrict__ out) {
    int lane = threadIdx.x & 31;
    int w    = threadIdx.x >> 5;
    int m0   = blockIdx.x * 64 + w * 16;
    int c0   = blockIdx.y * 32;

    v8f acc[2];
#pragma unroll
    for (int t = 0; t < 2; ++t) acc[t] = zero8f();

#pragma unroll
    for (int kk = 0; kk < EDIM; kk += 32) {
        v16h af = load_a_frag(z, m0, kk, EDIM, lane);
#pragma unroll
        for (int t = 0; t < 2; ++t) {
            v16h bf = load_b_frag(wt, c0 + t * 16, kk, EDIM, lane);
            acc[t] = wmma16(af, bf, acc[t]);
        }
    }

    int rbase = (lane >> 4) * 8;
#pragma unroll
    for (int t = 0; t < 2; ++t) {
        int   c = c0 + t * 16 + (lane & 15);
        float b = bias[c];
#pragma unroll
        for (int r = 0; r < 8; ++r) {
            size_t m = (size_t)(m0 + rbase + r);
            out[m * EDIM + c] = (_Float16)sigmoidf_(acc[t][r] + b);
        }
    }
}

// ---------------------------------------------------------------------------
// Triangle einsum: upd[i][j][c] = sum_k aT[c][i][k] * bT[c][j][k]
// 128 independent 512x512x512 NT GEMMs.
// Grid (8, 4, 128), block 128 (4 waves). WG tile 64(i) x 128(j) per channel.
// Per k-step (32): async-DMA A(64x32) + B(128x32) tiles into LDS (double
// buffered), each wave runs a 32x64 tile = 8 WMMAs from LDS fragments.
// ---------------------------------------------------------------------------
__global__ void tri_einsum_kernel(const _Float16* __restrict__ aT,
                                  const _Float16* __restrict__ bT,
                                  float* __restrict__ upd) {
    __shared__ __align__(64) _Float16 smA[2][64 * 32];    //  4 KB per buffer
    __shared__ __align__(64) _Float16 smB[2][128 * 32];   //  8 KB per buffer

    int tid  = threadIdx.x;
    int lane = tid & 31;
    int w    = tid >> 5;
    int c    = blockIdx.z;
    int i0g  = blockIdx.x * 64;
    int j0g  = blockIdx.y * 128;
    int iw   = (w & 1) * 32;     // wave's i offset inside the 64-row A tile
    int jw   = (w >> 1) * 64;    // wave's j offset inside the 128-row B tile

    const _Float16* A = aT + ((size_t)c << 18);
    const _Float16* B = bT + ((size_t)c << 18);

    v8f acc[2][4];
#pragma unroll
    for (int a = 0; a < 2; ++a)
#pragma unroll
        for (int b = 0; b < 4; ++b) acc[a][b] = zero8f();

    // ---- tile copy: A = 256 16B chunks (2/thread), B = 512 (4/thread) ----
    auto copy_tiles = [&](int kk, _Float16* dA, _Float16* dB) {
#pragma unroll
        for (int s = 0; s < 2; ++s) {
            int q   = tid + s * 128;
            int row = q >> 2;
            int col = (q & 3) * 8;
            async_copy16(A + (size_t)(i0g + row) * NDIM + kk + col,
                         dA + row * 32 + col);
        }
#pragma unroll
        for (int s = 0; s < 4; ++s) {
            int q   = tid + s * 128;
            int row = q >> 2;
            int col = (q & 3) * 8;
            async_copy16(B + (size_t)(j0g + row) * NDIM + kk + col,
                         dB + row * 32 + col);
        }
    };

    copy_tiles(0, smA[0], smB[0]);

    for (int s = 0; s < NDIM / 32; ++s) {
        int p = s & 1;
        wait_async_zero();        // this wave's copies into buffer p complete
        __syncthreads();          // all waves' copies done; prev reads retired
        if (s + 1 < NDIM / 32)    // overlap next DMA with this step's WMMAs
            copy_tiles((s + 1) * 32, smA[p ^ 1], smB[p ^ 1]);

        v16h a0 = load_a_frag(smA[p], iw,      0, 32, lane);
        v16h a1 = load_a_frag(smA[p], iw + 16, 0, 32, lane);
#pragma unroll
        for (int tn = 0; tn < 4; ++tn) {
            v16h bf = load_b_frag(smB[p], jw + tn * 16, 0, 32, lane);
            acc[0][tn] = wmma16(a0, bf, acc[0][tn]);
            acc[1][tn] = wmma16(a1, bf, acc[1][tn]);
        }
    }

    int rbase = (lane >> 4) * 8;
#pragma unroll
    for (int tm = 0; tm < 2; ++tm) {
#pragma unroll
        for (int tn = 0; tn < 4; ++tn) {
            int j = j0g + jw + tn * 16 + (lane & 15);
#pragma unroll
            for (int r = 0; r < 8; ++r) {
                int i = i0g + iw + tm * 16 + rbase + r;
                upd[((((size_t)i << 9) + (size_t)j) << 7) + c] = acc[tm][tn][r];
            }
        }
    }
}

// ---------------------------------------------------------------------------
// Final: out[m][e] = g[m][e] * (u_hat[m] @ lu_w + lu_b)[e]
// ---------------------------------------------------------------------------
__global__ void final_kernel(const _Float16* __restrict__ u,
                             const _Float16* __restrict__ luwt,
                             const float*    __restrict__ lub,
                             const _Float16* __restrict__ g,
                             float* __restrict__ out) {
    int lane = threadIdx.x & 31;
    int w    = threadIdx.x >> 5;
    int m0   = blockIdx.x * 64 + w * 16;
    int e0   = blockIdx.y * 64;

    v8f acc[4];
#pragma unroll
    for (int t = 0; t < 4; ++t) acc[t] = zero8f();

#pragma unroll
    for (int kk = 0; kk < EDIM; kk += 32) {
        v16h af = load_a_frag(u, m0, kk, EDIM, lane);
#pragma unroll
        for (int t = 0; t < 4; ++t) {
            v16h bf = load_b_frag(luwt, e0 + t * 16, kk, EDIM, lane);
            acc[t] = wmma16(af, bf, acc[t]);
        }
    }

    int rbase = (lane >> 4) * 8;
#pragma unroll
    for (int t = 0; t < 4; ++t) {
        int   e = e0 + t * 16 + (lane & 15);
        float b = lub[e];
#pragma unroll
        for (int r = 0; r < 8; ++r) {
            size_t m  = (size_t)(m0 + rbase + r);
            float  gv = (float)g[m * EDIM + e];
            out[m * EDIM + e] = (acc[t][r] + b) * gv;
        }
    }
}

// ---------------------------------------------------------------------------
// Host-side orchestration.
// Workspace layout (bytes):
//   [0,       64MB)  z_f16 [M][128]   (reused later as u_hat)
//   [64MB,   128MB)  aT    [128][512][512] f16
//   [128MB,  192MB)  bT    [128][512][512] f16
//   [192MB,  256MB)  g     [M][128] f16
//   [256MB,  384MB)  upd   [512][512][128] f32
//   [384MB,  ...  )  f16 transposed weights (6 x 32KB)
// ---------------------------------------------------------------------------
extern "C" void kernel_launch(void* const* d_in, const int* in_sizes, int n_in,
                              void* d_out, int out_size, void* d_ws, size_t ws_size,
                              hipStream_t stream) {
    const float* pair      = (const float*)d_in[0];
    const float* ln_pair_g = (const float*)d_in[1];
    const float* ln_pair_b = (const float*)d_in[2];
    const float* ga_w = (const float*)d_in[3];
    const float* ga_b = (const float*)d_in[4];
    const float* la_w = (const float*)d_in[5];
    const float* la_b = (const float*)d_in[6];
    const float* gb_w = (const float*)d_in[7];
    const float* gb_b = (const float*)d_in[8];
    const float* lb_w = (const float*)d_in[9];
    const float* lb_b = (const float*)d_in[10];
    const float* g_w  = (const float*)d_in[11];
    const float* g_b  = (const float*)d_in[12];
    const float* ln_up_g = (const float*)d_in[13];
    const float* ln_up_b = (const float*)d_in[14];
    const float* lu_w = (const float*)d_in[15];
    const float* lu_b = (const float*)d_in[16];

    char* ws = (char*)d_ws;
    const size_t SZ = 67108864ull;  // 64 MB
    _Float16* z    = (_Float16*)(ws + 0);        // also u_hat later
    _Float16* aT   = (_Float16*)(ws + SZ);
    _Float16* bT   = (_Float16*)(ws + 2 * SZ);
    _Float16* gbuf = (_Float16*)(ws + 3 * SZ);
    float*    upd  = (float*)   (ws + 4 * SZ);
    _Float16* wts  = (_Float16*)(ws + 6 * SZ);
    _Float16* wt_ga = wts;
    _Float16* wt_la = wts + 16384;
    _Float16* wt_gb = wts + 2 * 16384;
    _Float16* wt_lb = wts + 3 * 16384;
    _Float16* wt_g  = wts + 4 * 16384;
    _Float16* wt_lu = wts + 5 * 16384;

    // Weight transpose + f16 conversion
    wprep_kernel<<<64, 256, 0, stream>>>(ga_w, wt_ga);
    wprep_kernel<<<64, 256, 0, stream>>>(la_w, wt_la);
    wprep_kernel<<<64, 256, 0, stream>>>(gb_w, wt_gb);
    wprep_kernel<<<64, 256, 0, stream>>>(lb_w, wt_lb);
    wprep_kernel<<<64, 256, 0, stream>>>(g_w,  wt_g);
    wprep_kernel<<<64, 256, 0, stream>>>(lu_w, wt_lu);

    // LayerNorm of pair representation -> z (f16)
    ln_rows_kernel<<<MROWS / 8, 256, 0, stream>>>(pair, ln_pair_g, ln_pair_b, z, MROWS);

    // Projections (WMMA)
    dim3 gp(MROWS / 64, EDIM / 32);
    proj_pair_kernel<<<gp, 128, 0, stream>>>(z, wt_ga, ga_b, wt_la, la_b, aT);
    proj_pair_kernel<<<gp, 128, 0, stream>>>(z, wt_gb, gb_b, wt_lb, lb_b, bT);
    proj_g_kernel<<<gp, 128, 0, stream>>>(z, wt_g, g_b, gbuf);

    // Triangle einsum: 128 batched 512^3 NT GEMMs (WMMA + async LDS DMA)
    dim3 ge(NDIM / 64, NDIM / 128, EDIM);
    tri_einsum_kernel<<<ge, 128, 0, stream>>>(aT, bT, upd);

    // LayerNorm over channels of update -> u_hat (reuse z buffer)
    ln_rows_kernel<<<MROWS / 8, 256, 0, stream>>>(upd, ln_up_g, ln_up_b, z, MROWS);

    // Final projection, gated -> out (WMMA)
    dim3 gf(MROWS / 64, EDIM / 64);
    final_kernel<<<gf, 128, 0, stream>>>(z, wt_lu, lu_b, gbuf, (float*)d_out);
}
